// Encoder_67980742361360
// MI455X (gfx1250) — matrix operand order repaired and checked by reading, not compiled, not run on hardware
//
#include <hip/hip_runtime.h>
#include <hip/hip_bf16.h>

// Problem dims (fixed by the reference)
#define S_LEN  1024
#define DMODEL 1024
#define NHEAD  16
#define DHEAD  64
#define DFFN   4096
#define NLAYER 4
#define BATCH  4
#define MROWS  (BATCH * S_LEN)   // 4096 token rows

typedef __attribute__((ext_vector_type(16))) _Float16 v16h;
typedef __attribute__((ext_vector_type(8)))  float    v8f;
typedef __attribute__((ext_vector_type(4)))  unsigned int v4u;
typedef __attribute__((ext_vector_type(8)))  int      v8i;
typedef __attribute__((ext_vector_type(4)))  int      v4i;

#ifndef __has_builtin
#define __has_builtin(x) 0
#endif
#if defined(__HIP_DEVICE_COMPILE__) &&                              \
    __has_builtin(__builtin_amdgcn_tensor_load_to_lds) &&           \
    __has_builtin(__builtin_amdgcn_s_wait_tensorcnt)
#define HAVE_TDM 1
#else
#define HAVE_TDM 0
#endif

union Frag16 { v16h v; uint4 q[2]; };

// ---------------------------------------------------------------------------
// WMMA fragment loaders (wave32; A/B use this code's contiguous k order, not the ISA table order)
// A (16x32 f16): lane m = lane&15; hi = lane>>4; half h -> K = hi*16 + h
//                => 16 contiguous halves per lane.
// B (32x16 f16): lane n = lane&15; K = hi*16 + h => 16 contiguous halves.
// C/D (16x16 f32): row m = r + hi*8, col n = lane&15.
// ---------------------------------------------------------------------------
__device__ inline v16h load_a_frag(const _Float16* p, int ld, int kbase) {
  int lane = threadIdx.x & 31;
  const _Float16* row = p + (lane & 15) * ld + kbase + (lane >> 4) * 16;
  Frag16 f;
  f.q[0] = *(const uint4*)(row);
  f.q[1] = *(const uint4*)(row + 8);
  return f.v;
}

__device__ inline v16h load_b_frag(const _Float16* p, int ld, int nbase, int kbase) {
  int lane = threadIdx.x & 31;
  const _Float16* row = p + (nbase + (lane & 15)) * ld + kbase + (lane >> 4) * 16;
  Frag16 f;
  f.q[0] = *(const uint4*)(row);
  f.q[1] = *(const uint4*)(row + 8);
  return f.v;
}

__device__ inline v8f wmma_f16(v16h a, v16h b, v8f c) {
  return __builtin_amdgcn_wmma_f32_16x16x32_f16(false, a, false, b, (short)0, c,
                                                false, false);
}

#if HAVE_TDM
// TDM: async-load a 2D f16 tile (tile_dim0=32 elems per row, tile_dim1=128
// rows, row stride = K elems) from global into LDS. Descriptor per ISA §8.
__device__ inline void tdm_load_a_tile(const _Float16* gsrc, unsigned lds_off,
                                       int K) {
  unsigned long long ga = (unsigned long long)(uintptr_t)gsrc;
  v4u g0;
  g0.x = 1u;                                    // count=1, user mode
  g0.y = lds_off;                               // lds_addr (bytes)
  g0.z = (unsigned)(ga & 0xffffffffu);          // global_addr[31:0]
  g0.w = (unsigned)((ga >> 32) & 0x01ffffffu)   // global_addr[56:32]
         | (2u << 30);                          // type = 2 ("image")
  v8i g1;
  g1[0] = (1 << 16);                            // data_size=1 -> 2 bytes
  g1[1] = (int)(((unsigned)K & 0xffffu) << 16); // tensor_dim0[15:0]
  g1[2] = (int)(((unsigned)K >> 16) & 0xffffu)  // tensor_dim0[31:16]
        | (int)(((unsigned)MROWS & 0xffffu) << 16); // tensor_dim1[15:0]
  g1[3] = (int)(((unsigned)MROWS >> 16) & 0xffffu)  // tensor_dim1[31:16]
        | (32 << 16);                           // tile_dim0 = 32 elems
  g1[4] = 128;                                  // tile_dim1=128, tile_dim2=0
  g1[5] = (int)K;                               // tensor_dim0_stride[31:0]
  g1[6] = 0;                                    // stride hi / dim1_stride lo
  g1[7] = 0;
  v4i z4 = {0, 0, 0, 0};
#if defined(__clang_major__) && (__clang_major__ >= 23)
  v8i z8 = {0, 0, 0, 0, 0, 0, 0, 0};
  __builtin_amdgcn_tensor_load_to_lds(g0, g1, z4, z4, z8, 0);
#else
  __builtin_amdgcn_tensor_load_to_lds(g0, g1, z4, z4, 0);
#endif
}
#endif

// ---------------------------------------------------------------------------
// Tiled WMMA GEMM: out[M,N] = A[M,K](f16) * W[K,N](f32->f16) + bias (+resid)
// Block = 256 threads (8 waves), 128x128 tile; wave grid 2(M) x 4(N),
// each wave: 4x2 tiles of 16x16, K stepped by 32. Weights reg-prefetched one
// chunk ahead; A tile comes via the Tensor Data Mover when available.
// ---------------------------------------------------------------------------
template <int OUT16, int RELU, int RESID>
__global__ __launch_bounds__(256) void gemm_wmma_kernel(
    const _Float16* __restrict__ A, const float* __restrict__ W,
    const float* __restrict__ bias, const float* __restrict__ resid,
    _Float16* __restrict__ out16, float* __restrict__ out32,
    int M, int N, int K) {
  __shared__ _Float16 As[128 * 32];   // [m][k]
  __shared__ _Float16 Wt[128 * 32];   // [n][k] (transposed for B frags)
  const int t  = threadIdx.x;
  const int n0 = blockIdx.x * 128;
  const int m0 = blockIdx.y * 128;
  const int w  = t >> 5;
  const int wm = w >> 2, wn = w & 3;

  const int kk = t >> 3;              // W stager: k row 0..31
  const int nb = (t & 7) * 16;        //           n seg (16 floats)
  const int ar = t >> 1;              // A stager: row 0..127
  const int aseg = (t & 1) * 16;      //           k seg (16 halves)

  v8f acc[4][2] = {};

  // prefetch chunk 0 into registers
  float wreg[16];
  {
    const float* wp = W + (size_t)kk * N + n0 + nb;
#pragma unroll
    for (int j = 0; j < 16; ++j) wreg[j] = wp[j];
  }
#if !HAVE_TDM
  uint4 areg[2];
  {
    const uint4* ap = (const uint4*)(A + (size_t)(m0 + ar) * K + aseg);
    areg[0] = ap[0];
    areg[1] = ap[1];
  }
#endif

  for (int kc = 0; kc < K; kc += 32) {
    __syncthreads();  // previous chunk's consumers done
    {  // W tile: registers -> LDS, f32 -> f16, transposed [n][k]
#pragma unroll
      for (int j = 0; j < 16; ++j) Wt[(nb + j) * 32 + kk] = (_Float16)wreg[j];
    }
#if HAVE_TDM
    if (w == 0)  // one TDM descriptor per block: A tile 128x32 f16 -> LDS
      tdm_load_a_tile(A + (size_t)m0 * K + kc, (unsigned)(uintptr_t)As, K);
#else
    {
      uint4* dst = (uint4*)(As + ar * 32 + aseg);
      dst[0] = areg[0];
      dst[1] = areg[1];
    }
#endif
    if (kc + 32 < K) {  // issue next chunk's global loads (in flight @ WMMA)
      const float* wp = W + (size_t)(kc + 32 + kk) * N + n0 + nb;
#pragma unroll
      for (int j = 0; j < 16; ++j) wreg[j] = wp[j];
#if !HAVE_TDM
      const uint4* ap =
          (const uint4*)(A + (size_t)(m0 + ar) * K + kc + 32 + aseg);
      areg[0] = ap[0];
      areg[1] = ap[1];
#endif
    }
#if HAVE_TDM
    if (w == 0) __builtin_amdgcn_s_wait_tensorcnt((short)0);
#endif
    __syncthreads();  // tiles visible to all waves

    v16h af[4], bf[2];
#pragma unroll
    for (int mt = 0; mt < 4; ++mt)
      af[mt] = load_a_frag(As + (wm * 64 + mt * 16) * 32, 32, 0);
#pragma unroll
    for (int nt = 0; nt < 2; ++nt)
      bf[nt] = load_b_frag(Wt, 32, wn * 32 + nt * 16, 0);
#pragma unroll
    for (int mt = 0; mt < 4; ++mt)
#pragma unroll
      for (int nt = 0; nt < 2; ++nt)
        acc[mt][nt] = wmma_f16(af[mt], bf[nt], acc[mt][nt]);
  }

  const int lane = t & 31;
  const int ln = lane & 15, hi = lane >> 4;
#pragma unroll
  for (int mt = 0; mt < 4; ++mt) {
#pragma unroll
    for (int nt = 0; nt < 2; ++nt) {
      int n = n0 + wn * 32 + nt * 16 + ln;
      float bv = bias[n];
#pragma unroll
      for (int r = 0; r < 8; ++r) {
        int m = m0 + wm * 64 + mt * 16 + hi * 8 + r;
        float v = acc[mt][nt][r] + bv;
        if (RELU) v = v > 0.f ? v : 0.f;
        size_t idx = (size_t)m * N + n;
        if (RESID) v += resid[idx];
        if (OUT16) out16[idx] = (_Float16)v;
        else       out32[idx] = v;
      }
    }
  }
}

// ---------------------------------------------------------------------------
// Flash attention + role gate. Q/K/V/R are f16 [B,S,H*DH] ('bihd').
// Block = 128 threads = 4 waves, all on the same (b,h); wave handles a
// 16-query tile; loop over 32-key blocks with online softmax. The running
// softmax denominator is accumulated with an extra WMMA against a ones
// matrix (same rescale recurrence as O), removing the cross-lane sum.
// out = (softmax(QK^T/sqrt(d), mask) @ V) * R   (f16)
// ---------------------------------------------------------------------------
__global__ __launch_bounds__(128) void attn_wmma_kernel(
    const _Float16* __restrict__ Q, const _Float16* __restrict__ Km,
    const _Float16* __restrict__ V, const _Float16* __restrict__ R,
    const int* __restrict__ mask, _Float16* __restrict__ out) {
  __shared__ _Float16 Vt[DHEAD * 32];    // [d][key] for B frags
  __shared__ _Float16 Ps[4 * 16 * 32];   // per-wave P tile (C->A relayout)
  const int t = threadIdx.x;
  const int wave = t >> 5;
  const int lane = t & 31;
  const int ln = lane & 15, hi = lane >> 4;
  const int qb = blockIdx.x, h = blockIdx.y, b = blockIdx.z;
  const int q0 = qb * 64 + wave * 16;
  const float scale = 0.125f;  // 1/sqrt(64)

  const _Float16* Qbase = Q + (size_t)(b * S_LEN + q0) * DMODEL + h * DHEAD;
  v16h aq[2];
  aq[0] = load_a_frag(Qbase, DMODEL, 0);
  aq[1] = load_a_frag(Qbase, DMODEL, 32);

  v16h ones;
#pragma unroll
  for (int i = 0; i < 16; ++i) ones[i] = (_Float16)1.0f;

  float rm[8];
  v8f o[4] = {};
  v8f osum = {};
#pragma unroll
  for (int r = 0; r < 8; ++r) rm[r] = -3.0e38f;

  // this thread's V-staging source (advances 32 keys per block)
  const int skey = t & 31, dseg = (t >> 5) * 16;
  const _Float16* vstage =
      V + (size_t)(b * S_LEN + skey) * DMODEL + h * DHEAD + dseg;

  for (int kb = 0; kb < S_LEN / 32; ++kb) {
    __syncthreads();
    {  // stage V tile (32 keys x 64 d) transposed -> Vt[d][key]
      const _Float16* vp = vstage + (size_t)kb * 32 * DMODEL;
#pragma unroll
      for (int j = 0; j < 16; ++j) Vt[(dseg + j) * 32 + skey] = vp[j];
      if (kb + 1 < S_LEN / 32)  // global_prefetch next V tile
        __builtin_prefetch(vp + (size_t)32 * DMODEL, 0, 1);
    }
    __syncthreads();

    // S = Q K^T  (16 x 32), K frags streamed straight from global
    float sv[2][8];
#pragma unroll
    for (int nt = 0; nt < 2; ++nt) {
      v8f s = {};
      int key = kb * 32 + nt * 16 + ln;
#pragma unroll
      for (int kf = 0; kf < 2; ++kf) {
        const _Float16* kp = Km + (size_t)(b * S_LEN + key) * DMODEL +
                             h * DHEAD + kf * 32 + hi * 16;
        Frag16 f;
        f.q[0] = *(const uint4*)kp;
        f.q[1] = *(const uint4*)(kp + 8);
        s = wmma_f16(aq[kf], f.v, s);
      }
      int mk = mask[b * S_LEN + key];
#pragma unroll
      for (int r = 0; r < 8; ++r) {
        float x = s[r] * scale;
        sv[nt][r] = mk ? x : -1.0e10f;
      }
    }

    // row max over 32 cols (across the 16-lane N group)
    float cm[8];
#pragma unroll
    for (int r = 0; r < 8; ++r) cm[r] = fmaxf(sv[0][r], sv[1][r]);
#pragma unroll
    for (int msk = 1; msk <= 8; msk <<= 1)
#pragma unroll
      for (int r = 0; r < 8; ++r)
        cm[r] = fmaxf(cm[r], __shfl_xor(cm[r], msk, 32));

    float pv[2][8];
#pragma unroll
    for (int r = 0; r < 8; ++r) {
      float nm = fmaxf(rm[r], cm[r]);
      float corr = __expf(rm[r] - nm);
#pragma unroll
      for (int dt = 0; dt < 4; ++dt) o[dt][r] *= corr;
      osum[r] *= corr;
      rm[r] = nm;
      pv[0][r] = __expf(sv[0][r] - nm);
      pv[1][r] = __expf(sv[1][r] - nm);
    }

    // P tile C-layout -> LDS -> A-layout
    _Float16* pw = Ps + wave * 512;
#pragma unroll
    for (int nt = 0; nt < 2; ++nt)
#pragma unroll
      for (int r = 0; r < 8; ++r)
        pw[(r + hi * 8) * 32 + nt * 16 + ln] = (_Float16)pv[nt][r];
    __syncthreads();

    v16h ap = load_a_frag(pw, 32, 0);
    osum = wmma_f16(ap, ones, osum);  // row sums, replicated across lanes
#pragma unroll
    for (int dt = 0; dt < 4; ++dt) {
      v16h bv = load_b_frag(Vt, 32, dt * 16, 0);
      o[dt] = wmma_f16(ap, bv, o[dt]);
    }
  }

  // normalize, role-gate, store
#pragma unroll
  for (int dt = 0; dt < 4; ++dt) {
#pragma unroll
    for (int r = 0; r < 8; ++r) {
      int s = q0 + hi * 8 + r;
      int d = dt * 16 + ln;
      size_t idx = (size_t)(b * S_LEN + s) * DMODEL + h * DHEAD + d;
      float rg = (float)R[idx];
      out[idx] = (_Float16)(o[dt][r] / osum[r] * rg);
    }
  }
}

// ---------------------------------------------------------------------------
// LayerNorm over D=1024; one row per 256-thread block; f16 or f32 output.
// ---------------------------------------------------------------------------
__global__ __launch_bounds__(256) void ln_kernel(
    const float* __restrict__ x, const float* __restrict__ gam,
    const float* __restrict__ bet, _Float16* __restrict__ y16,
    float* __restrict__ y32) {
  __shared__ float red[256];
  const int t = threadIdx.x;
  const size_t row = blockIdx.x;
  const float4 v = ((const float4*)(x + row * DMODEL))[t];
  red[t] = v.x + v.y + v.z + v.w;
  __syncthreads();
  for (int off = 128; off > 0; off >>= 1) {
    if (t < off) red[t] += red[t + off];
    __syncthreads();
  }
  float mean = red[0] * (1.f / DMODEL);
  __syncthreads();
  float dx = v.x - mean, dy = v.y - mean, dz = v.z - mean, dw = v.w - mean;
  red[t] = dx * dx + dy * dy + dz * dz + dw * dw;
  __syncthreads();
  for (int off = 128; off > 0; off >>= 1) {
    if (t < off) red[t] += red[t + off];
    __syncthreads();
  }
  float inv = rsqrtf(red[0] * (1.f / DMODEL) + 1e-5f);
  float4 g4 = ((const float4*)gam)[t];
  float4 b4 = ((const float4*)bet)[t];
  float o0 = dx * inv * g4.x + b4.x;
  float o1 = dy * inv * g4.y + b4.y;
  float o2 = dz * inv * g4.z + b4.z;
  float o3 = dw * inv * g4.w + b4.w;
  if (y16) {
    _Float16* yp = y16 + row * DMODEL + t * 4;
    yp[0] = (_Float16)o0; yp[1] = (_Float16)o1;
    yp[2] = (_Float16)o2; yp[3] = (_Float16)o3;
  } else {
    ((float4*)(y32 + row * DMODEL))[t] = make_float4(o0, o1, o2, o3);
  }
}

// ---------------------------------------------------------------------------
extern "C" void kernel_launch(void* const* d_in, const int* in_sizes, int n_in,
                              void* d_out, int out_size, void* d_ws,
                              size_t ws_size, hipStream_t stream) {
  (void)in_sizes; (void)n_in; (void)out_size; (void)ws_size;
  const float* src  = (const float*)d_in[0];
  const int*   mask = (const int*)d_in[1];
  const float* ln1g = (const float*)d_in[2];
  const float* ln1b = (const float*)d_in[3];
  const float* Wq = (const float*)d_in[4];
  const float* bq = (const float*)d_in[5];
  const float* Wk = (const float*)d_in[6];
  const float* bk = (const float*)d_in[7];
  const float* Wv = (const float*)d_in[8];
  const float* bv = (const float*)d_in[9];
  const float* Wr = (const float*)d_in[10];
  const float* br = (const float*)d_in[11];
  const float* Wo = (const float*)d_in[12];
  const float* bo = (const float*)d_in[13];
  const float* ln2g = (const float*)d_in[14];
  const float* ln2b = (const float*)d_in[15];
  const float* W1 = (const float*)d_in[16];
  const float* b1 = (const float*)d_in[17];
  const float* W2 = (const float*)d_in[18];
  const float* b2 = (const float*)d_in[19];
  const float* ln3g = (const float*)d_in[20];
  const float* ln3b = (const float*)d_in[21];

  const size_t MB = 1ull << 20;
  char* ws = (char*)d_ws;
  _Float16* z16   = (_Float16*)(ws);             //  8 MB  LN1 out
  _Float16* q16   = (_Float16*)(ws + 8 * MB);    //  8 MB
  _Float16* k16   = (_Float16*)(ws + 16 * MB);   //  8 MB
  _Float16* vv16  = (_Float16*)(ws + 24 * MB);   //  8 MB
  _Float16* r16   = (_Float16*)(ws + 32 * MB);   //  8 MB
  _Float16* att16 = (_Float16*)(ws + 40 * MB);   //  8 MB  gated attn out
  float*    mha   = (float*)(ws + 48 * MB);      // 16 MB  x + Wo(attn)
  _Float16* z2    = (_Float16*)(ws + 64 * MB);   //  8 MB  LN2 out
  _Float16* ff1   = (_Float16*)(ws + 72 * MB);   // 32 MB  relu(z2 W1)
  float*    ybuf  = (float*)(ws + 104 * MB);     // 16 MB  mha + FFN
  float*    xbuf  = (float*)(ws + 120 * MB);     // 16 MB  residual stream

  const dim3 gN1024(DMODEL / 128, MROWS / 128);  // (8, 32)
  const dim3 gN4096(DFFN / 128, MROWS / 128);    // (32, 32)
  const dim3 gAttn(S_LEN / 64, NHEAD, BATCH);    // (16, 16, 4)

  for (int l = 0; l < NLAYER; ++l) {
    const float* xin = (l == 0) ? src : xbuf;
    const size_t oD  = (size_t)l * DMODEL;
    const size_t oDD = (size_t)l * DMODEL * DMODEL;
    const size_t oDF = (size_t)l * DMODEL * DFFN;

    // z = LN1(x) -> f16
    ln_kernel<<<MROWS, 256, 0, stream>>>(xin, ln1g + oD, ln1b + oD, z16, nullptr);

    // Q,K,V,R projections
    gemm_wmma_kernel<1, 0, 0><<<gN1024, 256, 0, stream>>>(
        z16, Wq + oDD, bq + oD, nullptr, q16, nullptr, MROWS, DMODEL, DMODEL);
    gemm_wmma_kernel<1, 0, 0><<<gN1024, 256, 0, stream>>>(
        z16, Wk + oDD, bk + oD, nullptr, k16, nullptr, MROWS, DMODEL, DMODEL);
    gemm_wmma_kernel<1, 0, 0><<<gN1024, 256, 0, stream>>>(
        z16, Wv + oDD, bv + oD, nullptr, vv16, nullptr, MROWS, DMODEL, DMODEL);
    gemm_wmma_kernel<1, 0, 0><<<gN1024, 256, 0, stream>>>(
        z16, Wr + oDD, br + oD, nullptr, r16, nullptr, MROWS, DMODEL, DMODEL);

    // flash attention + role gate
    attn_wmma_kernel<<<gAttn, 128, 0, stream>>>(q16, k16, vv16, r16, mask, att16);

    // mha_res = x + attn @ Wo + bo
    gemm_wmma_kernel<0, 0, 1><<<gN1024, 256, 0, stream>>>(
        att16, Wo + oDD, bo + oD, xin, nullptr, mha, MROWS, DMODEL, DMODEL);

    // z2 = LN2(mha_res) -> f16
    ln_kernel<<<MROWS, 256, 0, stream>>>(mha, ln2g + oD, ln2b + oD, z2, nullptr);

    // ff1 = relu(z2 @ W1 + b1) -> f16
    gemm_wmma_kernel<1, 1, 0><<<gN4096, 256, 0, stream>>>(
        z2, W1 + oDF, b1 + (size_t)l * DFFN, nullptr, ff1, nullptr,
        MROWS, DFFN, DMODEL);

    // y = mha_res + ff1 @ W2 + b2
    gemm_wmma_kernel<0, 0, 1><<<gN1024, 256, 0, stream>>>(
        ff1, W2 + oDF, b2 + oD, mha, nullptr, ybuf, MROWS, DMODEL, DFFN);

    // x' = LN3(y)  (final layer -> d_out)
    float* ln3out = (l == NLAYER - 1) ? (float*)d_out : xbuf;
    ln_kernel<<<MROWS, 256, 0, stream>>>(ybuf, ln3g + oD, ln3b + oD, nullptr, ln3out);
  }
}
